// SimCLRLoss_28630251995607
// MI455X (gfx1250) — matrix operand order repaired and checked by reading, not compiled, not run on hardware
//
#include <hip/hip_runtime.h>
#include <math.h>

typedef __attribute__((ext_vector_type(16))) _Float16 v16h;
typedef __attribute__((ext_vector_type(8)))  _Float16 v8h;
typedef __attribute__((ext_vector_type(8)))  float    v8f;
typedef __attribute__((ext_vector_type(4)))  float    v4f;

#define NPAIR 4096
#define DDIM  256
#define NROWS 8192              // 2N
#define EPSN  1e-8f
#define INV_T 2.0f              // 1 / 0.5
#define CSLC  4                 // column slices (grid.y)
#define NDT   (NROWS / 32)      // 256 double-tiles (32 cols each)
#define NDTS  (NDT / CSLC)      // 64 double-tiles per slice
#define BROWH 264               // padded LDS row: 264 halves = 528 B (bank-conflict free)

// ---------------- zero denom[8192] + acc[2] ----------------
__global__ void k_zero(float* p, int n) {
    int i = blockIdx.x * blockDim.x + threadIdx.x;
    if (i < n) p[i] = 0.0f;
}

// ---------------- normalize rows of [z1;z2] into f16 ----------------
__global__ void k_normalize(const float* __restrict__ z1,
                            const float* __restrict__ z2,
                            _Float16* __restrict__ fnh) {
    int wave = (blockIdx.x * blockDim.x + threadIdx.x) >> 5;
    int lane = threadIdx.x & 31;
    if (wave >= NROWS) return;
    const float* src = (wave < NPAIR) ? (z1 + (size_t)wave * DDIM)
                                      : (z2 + (size_t)(wave - NPAIR) * DDIM);
    v4f a = *(const v4f*)(src + lane * 8);
    v4f b = *(const v4f*)(src + lane * 8 + 4);
    float ss = a.x*a.x + a.y*a.y + a.z*a.z + a.w*a.w
             + b.x*b.x + b.y*b.y + b.z*b.z + b.w*b.w;
    #pragma unroll
    for (int m = 16; m >= 1; m >>= 1) ss += __shfl_xor(ss, m, 32);
    float inv = 1.0f / fmaxf(sqrtf(ss), EPSN);
    v8h h;
    h[0] = (_Float16)(a.x * inv); h[1] = (_Float16)(a.y * inv);
    h[2] = (_Float16)(a.z * inv); h[3] = (_Float16)(a.w * inv);
    h[4] = (_Float16)(b.x * inv); h[5] = (_Float16)(b.y * inv);
    h[6] = (_Float16)(b.z * inv); h[7] = (_Float16)(b.w * inv);
    *(v8h*)(fnh + (size_t)wave * DDIM + lane * 8) = h;
}

// ---------------- positive-pair cosine term (pure f32) ----------------
__global__ void k_pos(const float* __restrict__ z1,
                      const float* __restrict__ z2,
                      float* __restrict__ acc) {
    int wave = (blockIdx.x * blockDim.x + threadIdx.x) >> 5;
    int lane = threadIdx.x & 31;
    if (wave >= NPAIR) return;
    const float* p = z1 + (size_t)wave * DDIM + lane * 8;
    const float* q = z2 + (size_t)wave * DDIM + lane * 8;
    float s11 = 0.f, s22 = 0.f, s12 = 0.f;
    #pragma unroll
    for (int i = 0; i < 8; ++i) {
        float x = p[i], y = q[i];
        s11 += x * x; s22 += y * y; s12 += x * y;
    }
    #pragma unroll
    for (int m = 16; m >= 1; m >>= 1) {
        s11 += __shfl_xor(s11, m, 32);
        s22 += __shfl_xor(s22, m, 32);
        s12 += __shfl_xor(s12, m, 32);
    }
    if (lane == 0) {
        float pos = s12 / (fmaxf(sqrtf(s11), EPSN) * fmaxf(sqrtf(s22), EPSN));
        atomicAdd(&acc[1], pos);
    }
}

// async global -> LDS, 16 bytes per lane (ASYNCcnt-tracked)
__device__ __forceinline__ void async_ld16(unsigned lds, const _Float16* g) {
    asm volatile("global_load_async_to_lds_b128 %0, %1, off"
                 :: "v"(lds), "v"((unsigned long long)(uintptr_t)g) : "memory");
}

// issue the 8 per-thread 16B chunks that cooperatively fill one 16KB tile
__device__ __forceinline__ void fill_tile(const unsigned (&lofs)[8], const int (&gofs)[8],
                                          const _Float16* gtile) {
    #pragma unroll
    for (int i = 0; i < 8; ++i) async_ld16(lofs[i], gtile + gofs[i]);
}

// B fragment (32x16 f16) from LDS: lane = column, K {0..7,16..23} | {8..15,24..31} per lane group
__device__ __forceinline__ v16h ld_bfrag(const _Float16* bb, int l15, int halfSel,
                                         int j, int c) {
    const int k0 = c * 32 + halfSel * 8;
    const _Float16* p = bb + (j * 16 + l15) * BROWH + k0;
    v8h lo = *(const v8h*)p;
    v8h hi = *(const v8h*)(p + 16);
    v16h f;
    #pragma unroll
    for (int i = 0; i < 8; ++i) { f[i] = lo[i]; f[i + 8] = hi[i]; }
    return f;
}

// consume one 32-col tile: 16 WMMAs with depth-2 software-pipelined DS loads
__device__ __forceinline__ void consume_tile(const _Float16* bb, const v16h (&afrag)[8],
                                             int l15, int halfSel, int colBase,
                                             int rowBase, int rowOff, float (&rs)[8]) {
    v8f ca = {}, cb = {};
    v16h f0 = ld_bfrag(bb, l15, halfSel, 0, 0);
    v16h f1 = ld_bfrag(bb, l15, halfSel, 0, 1);
    #pragma unroll
    for (int q = 0; q < 16; ++q) {
        const int qn = q + 2;
        v16h f2 = (qn < 16) ? ld_bfrag(bb, l15, halfSel, (qn >> 3) & 1, qn & 7) : f0;
        const int c = q & 7;
        if (q < 8)
            ca = __builtin_amdgcn_wmma_f32_16x16x32_f16(
                     false, afrag[c], false, f0, (short)0, ca, false, false);
        else
            cb = __builtin_amdgcn_wmma_f32_16x16x32_f16(
                     false, afrag[c], false, f0, (short)0, cb, false, false);
        f0 = f1; f1 = f2;
    }
    #pragma unroll
    for (int j = 0; j < 2; ++j) {
        const v8f& cc = j ? cb : ca;
        const int col = colBase + j * 16 + l15;
        #pragma unroll
        for (int r = 0; r < 8; ++r) {
            int row = rowBase + r + rowOff;
            rs[r] += (row == col) ? 0.0f : __expf(cc[r] * INV_T);
        }
    }
}

// ---------------- Gram matrix + exp-sum, WMMA f16 + async-LDS staging ----
// grid = (128 row-blocks x 4 col slices), block = 4 waves (64 rows / block).
// 32-col B tiles (16 KB) double-buffered in LDS via async loads, shared by
// all 4 waves; per stage each wave runs 16 WMMAs against resident A frags.
__global__ void __launch_bounds__(128)
k_gram(const _Float16* __restrict__ fnh, float* __restrict__ denom) {
    __shared__ __align__(16) _Float16 sbuf[2][32 * BROWH];   // 2 x 16.5 KB

    const int tid     = threadIdx.x;
    const int lane    = tid & 31;
    const int wid     = tid >> 5;
    const int l15     = lane & 15;
    const int halfSel = (lane >> 4) & 1;
    const int rowBase = blockIdx.x * 64 + wid * 16;
    const int rowOff  = halfSel * 8;            // C/D: lanes 16-31 hold M = r + 8
    const int dt0     = blockIdx.y * NDTS;      // first double-tile of this slice

    // A fragments (16x32 f16), ISA 7.12.2 layout: K {0..7,16..23} | {8..15,24..31}
    const _Float16* arow = fnh + (size_t)(rowBase + l15) * DDIM;
    v16h afrag[8];
    #pragma unroll
    for (int c = 0; c < 8; ++c) {
        int k0 = c * 32 + halfSel * 8;
        v8h lo = *(const v8h*)(arow + k0);
        v8h hi = *(const v8h*)(arow + k0 + 16);
        #pragma unroll
        for (int i = 0; i < 8; ++i) { afrag[c][i] = lo[i]; afrag[c][i + 8] = hi[i]; }
    }

    // cooperative fill: 128 threads x 8 chunks of 16B cover a 16KB tile
    unsigned lofs0[8], lofs1[8];
    int gofs[8];
    #pragma unroll
    for (int i = 0; i < 8; ++i) {
        int c  = tid + 128 * i;
        int rr = c >> 5;        // row within tile (0..31)
        int oo = c & 31;        // 16B chunk within row (0..31)
        gofs[i]  = rr * DDIM + oo * 8;
        lofs0[i] = (unsigned)(uintptr_t)&sbuf[0][rr * BROWH + oo * 8];
        lofs1[i] = (unsigned)(uintptr_t)&sbuf[1][rr * BROWH + oo * 8];
    }

    float rs[8];
    #pragma unroll
    for (int r = 0; r < 8; ++r) rs[r] = 0.0f;

    // prologue: fill buffer 0 with first tile
    fill_tile(lofs0, gofs, fnh + (size_t)dt0 * 32 * DDIM);

    for (int t = 0; t < NDTS; t += 2) {
        // ---- stage A: consume buf0 (tile t), prefetch tile t+1 into buf1 ----
        fill_tile(lofs1, gofs, fnh + (size_t)(dt0 + t + 1) * 32 * DDIM);
        asm volatile("s_wait_asynccnt 8" ::: "memory");   // own tile-t chunks done
        __syncthreads();                                  // all waves' chunks done
        consume_tile(&sbuf[0][0], afrag, l15, halfSel, (dt0 + t) * 32,
                     rowBase, rowOff, rs);
        __syncthreads();                                  // buf0 free for refill

        // ---- stage B: consume buf1 (tile t+1), prefetch t+2 into buf0 ----
        const int tn = (t + 2 < NDTS) ? (t + 2) : (t + 1);   // clamp on last
        fill_tile(lofs0, gofs, fnh + (size_t)(dt0 + tn) * 32 * DDIM);
        asm volatile("s_wait_asynccnt 8" ::: "memory");
        __syncthreads();
        consume_tile(&sbuf[1][0], afrag, l15, halfSel, (dt0 + t + 1) * 32,
                     rowBase, rowOff, rs);
        __syncthreads();
    }

    // reduce each row-sum across the 16 lanes of each half-wave
    #pragma unroll
    for (int r = 0; r < 8; ++r) {
        #pragma unroll
        for (int m = 8; m >= 1; m >>= 1) rs[r] += __shfl_xor(rs[r], m, 32);
    }
    if (l15 == 0) {
        #pragma unroll
        for (int r = 0; r < 8; ++r)
            atomicAdd(&denom[rowBase + r + rowOff], rs[r]);
    }
}

// ---------------- sum of log(denom) ----------------
__global__ void k_logsum(const float* __restrict__ denom, float* __restrict__ acc) {
    int i    = blockIdx.x * blockDim.x + threadIdx.x;
    int lane = threadIdx.x & 31;
    float lg = (i < NROWS) ? __logf(denom[i]) : 0.0f;
    #pragma unroll
    for (int m = 16; m >= 1; m >>= 1) lg += __shfl_xor(lg, m, 32);
    if (lane == 0) atomicAdd(&acc[0], lg);
}

// ---------------- combine to scalar loss ----------------
// loss = mean(log denom) - mean(pos / T) = (acc0 - 2*INV_T*acc1) / 8192
__global__ void k_final(const float* __restrict__ acc, float* __restrict__ out) {
    if (threadIdx.x == 0 && blockIdx.x == 0)
        out[0] = (acc[0] - 2.0f * INV_T * acc[1]) * (1.0f / (float)NROWS);
}

extern "C" void kernel_launch(void* const* d_in, const int* in_sizes, int n_in,
                              void* d_out, int out_size, void* d_ws, size_t ws_size,
                              hipStream_t stream) {
    (void)in_sizes; (void)n_in; (void)out_size; (void)ws_size;
    const float* z1 = (const float*)d_in[0];
    const float* z2 = (const float*)d_in[1];
    float* out = (float*)d_out;

    _Float16* fnh = (_Float16*)d_ws;                                   // 4 MiB
    float* denom  = (float*)((char*)d_ws + (size_t)NROWS * DDIM * 2);  // 8192 floats
    float* acc    = denom + NROWS;                                     // 2 floats

    k_zero<<<(NROWS + 2 + 255) / 256, 256, 0, stream>>>(denom, NROWS + 2);
    k_normalize<<<NROWS / 8, 256, 0, stream>>>(z1, z2, fnh);     // 8 waves/block
    k_pos<<<NPAIR / 8, 256, 0, stream>>>(z1, z2, acc);
    dim3 grid(NROWS / 64, CSLC);                                 // 128 x 4 blocks
    k_gram<<<grid, 128, 0, stream>>>(fnh, denom);
    k_logsum<<<NROWS / 256, 256, 0, stream>>>(denom, acc);
    k_final<<<1, 32, 0, stream>>>(acc, out);
}
